// EquivariantUpdate_70351564309242
// MI455X (gfx1250) — matrix operand order repaired and checked by reading, not compiled, not run on hardware
//
#include <hip/hip_runtime.h>

typedef __attribute__((ext_vector_type(16))) __bf16 v16bf;
typedef __attribute__((ext_vector_type(8)))  float  v8f;

#define HDIM 128
#define INV_NORM (1.0f / 100.0f)
#define TILES_PER_WAVE 4

struct Pack32 { uint4 lo, hi; };

static __device__ __forceinline__ v16bf make_frag(uint4 lo, uint4 hi) {
  Pack32 p{lo, hi};
  return __builtin_bit_cast(v16bf, p);
}

// fp32 -> bf16, round to nearest even
static __device__ __forceinline__ unsigned short f2bf(float f) {
  unsigned int u = __float_as_uint(f);
  u += 0x7FFFu + ((u >> 16) & 1u);
  return (unsigned short)(u >> 16);
}

static __device__ __forceinline__ float silu(float v) {
  return v * __builtin_amdgcn_rcpf(1.0f + __expf(-v));
}

// ---------------------------------------------------------------------------
// Prep: convert W1[:, :256] and W2 to bf16 (row-major [out_feat][k])
// ---------------------------------------------------------------------------
__global__ void prep_weights(const float* __restrict__ W1, const float* __restrict__ W2,
                             unsigned short* __restrict__ w1a,
                             unsigned short* __restrict__ w1b,
                             unsigned short* __restrict__ w2b) {
  int t = blockIdx.x * blockDim.x + threadIdx.x;
  if (t >= HDIM * HDIM) return;
  int j = t >> 7, k = t & 127;
  w1a[t] = f2bf(W1[j * 257 + k]);
  w1b[t] = f2bf(W1[j * 257 + 128 + k]);
  w2b[t] = f2bf(W2[t]);
}

// Prep: h fp32 -> bf16, and out = coord
__global__ void prep_nodes(const float* __restrict__ h, const float* __restrict__ coord,
                           unsigned short* __restrict__ h16, float* __restrict__ out, int N) {
  int t = blockIdx.x * blockDim.x + threadIdx.x;
  if (t < N * 3) out[t] = coord[t];
  if (t < N * HDIM) h16[t] = f2bf(h[t]);
}

// ---------------------------------------------------------------------------
// Main edge kernel: weights resident in LDS, one wave32 per 16-edge tile,
// TILES_PER_WAVE tiles per wave. K-outer / j-inner WMMA order with 4
// independent accumulators to pipeline ds_load_b128 against v_wmma.
// ---------------------------------------------------------------------------
__global__ void __launch_bounds__(256) edge_kernel(
    const unsigned short* __restrict__ h16,      // [N,128] bf16
    const uint4* __restrict__ wpack,             // 96KB contiguous: w1a|w1b|w2b (bf16)
    const int* __restrict__ ei,                  // [2,E] indices
    const float* __restrict__ coord_diff,        // [E,3]
    const float* __restrict__ edge_attr,         // [E,1]
    const float* __restrict__ edge_mask,         // [E,1]
    const float* __restrict__ W1,                // [128,257] fp32 (for k=256 col)
    const float* __restrict__ b1,                // [128]
    const float* __restrict__ b2,                // [128]
    const float* __restrict__ W3,                // [128]
    float* __restrict__ out,                     // [N,3]
    int E) {
  __shared__ __align__(16) unsigned short wlds[3 * HDIM * HDIM];  // 96KB weights
  __shared__ __align__(16) unsigned short ldsx[8][16 * HDIM];     // 32KB x1 staging

  // ---- cooperative weight stage: global -> LDS, once per block ----
  {
    uint4* wl4 = reinterpret_cast<uint4*>(wlds);
    const int n4 = 3 * HDIM * HDIM / 8;  // 6144 uint4
#pragma unroll 4
    for (int i = threadIdx.x; i < n4; i += 256) wl4[i] = wpack[i];
  }
  __syncthreads();

  const int lane = threadIdx.x & 31;
  const int wv   = threadIdx.x >> 5;
  const int half = lane >> 4;   // 0 or 1
  const int n    = lane & 15;   // N-lane (B/C/D) or M-row (A)

  const unsigned short* w1a_l = wlds;
  const unsigned short* w1b_l = wlds + HDIM * HDIM;
  const unsigned short* w2b_l = wlds + 2 * HDIM * HDIM;

  // ---- hoist per-lane tile-invariant scalars ----
  float bj1[8], wc1[8], bj2[8], w3v[8];
#pragma unroll
  for (int j = 0; j < 8; ++j) {
    const int colf = j * 16 + n;
    bj1[j] = b1[colf];
    wc1[j] = W1[colf * 257 + 256];   // rank-1 edge_attr column of W1
    bj2[j] = b2[colf];
    w3v[j] = W3[colf];
  }

  const long long wave_id = (long long)blockIdx.x * (blockDim.x >> 5) + wv;

  for (int t = 0; t < TILES_PER_WAVE; ++t) {
    const long long e0 = (wave_id * TILES_PER_WAVE + t) * 16LL;
    if (e0 >= E) break;   // wave-uniform: EXEC stays all-ones inside

    // --- edge indices (clamped for a possible partial tail tile) ---
    long long ie = e0 + n; if (ie >= E) ie = E - 1;
    const int idxr = ei[ie];
    const int idxc = ei[(long long)E + ie];

    // --- A fragments: gathered h rows, 16-bit A layout (lane m; K 8+8 per half)
    v16bf ar[4], ac[4];
    {
      const uint4* pr = reinterpret_cast<const uint4*>(h16 + (long long)idxr * HDIM + half * 8);
      const uint4* pc = reinterpret_cast<const uint4*>(h16 + (long long)idxc * HDIM + half * 8);
#pragma unroll
      for (int s = 0; s < 4; ++s) {
        ar[s] = make_frag(pr[s * 4], pr[s * 4 + 2]);   // K = 32s+8h+[0,8) and +16
        ac[s] = make_frag(pc[s * 4], pc[s * 4 + 2]);
      }
    }

    // --- edge_attr values for the 8 edge rows this lane's C-VGPRs cover ---
    float ea[8];
#pragma unroll
    for (int r = 0; r < 8; ++r) {
      long long m = e0 + 8 * half + r; if (m >= E) m = E - 1;
      ea[r] = edge_attr[m];
    }

    // ================= GEMM1: x1 = silu([hr|hc|ea] @ W1^T + b1) =============
    // j-tiles in groups of 4: K-slice outer, 4 independent accumulators inner.
#pragma unroll
    for (int g = 0; g < 2; ++g) {
      v8f c[4];
#pragma unroll
      for (int jj = 0; jj < 4; ++jj) {
        const int j = g * 4 + jj;
#pragma unroll
        for (int r = 0; r < 8; ++r) c[jj][r] = __builtin_fmaf(ea[r], wc1[j], bj1[j]);
      }

#pragma unroll
      for (int s = 0; s < 4; ++s) {
#pragma unroll
        for (int jj = 0; jj < 4; ++jj) {
          const int colf = (g * 4 + jj) * 16 + n;
          const uint4* pb = reinterpret_cast<const uint4*>(w1a_l + colf * HDIM + half * 8);
          v16bf b = make_frag(pb[s * 4], pb[s * 4 + 2]);  // K = 32s+8h+[0,8) and +16
          c[jj] = __builtin_amdgcn_wmma_f32_16x16x32_bf16(false, ar[s], false, b,
                                                          (short)0, c[jj], false, false);
        }
      }
#pragma unroll
      for (int s = 0; s < 4; ++s) {
#pragma unroll
        for (int jj = 0; jj < 4; ++jj) {
          const int colf = (g * 4 + jj) * 16 + n;
          const uint4* pb = reinterpret_cast<const uint4*>(w1b_l + colf * HDIM + half * 8);
          v16bf b = make_frag(pb[s * 4], pb[s * 4 + 2]);
          c[jj] = __builtin_amdgcn_wmma_f32_16x16x32_bf16(false, ac[s], false, b,
                                                          (short)0, c[jj], false, false);
        }
      }
      // silu + stage to LDS in [edge][feature] order (bf16)
#pragma unroll
      for (int jj = 0; jj < 4; ++jj) {
        const int colf = (g * 4 + jj) * 16 + n;
#pragma unroll
        for (int r = 0; r < 8; ++r)
          ldsx[wv][(r + 8 * half) * HDIM + colf] = f2bf(silu(c[jj][r]));
      }
    }

    // Wave-private LDS region: drain DS ops before re-reading (CDNA5 wait)
    asm volatile("s_wait_dscnt 0" ::: "memory");

    // --- reload x1 as A fragments from LDS ---
    v16bf a2[4];
    {
      const uint4* pl = reinterpret_cast<const uint4*>(&ldsx[wv][n * HDIM + half * 8]);
#pragma unroll
      for (int s = 0; s < 4; ++s) a2[s] = make_frag(pl[s * 4], pl[s * 4 + 2]);
    }

    // ============ GEMM2 + fused W3 dot: phi = silu(x1@W2^T+b2) @ W3 ==========
    float p[8];
#pragma unroll
    for (int r = 0; r < 8; ++r) p[r] = 0.0f;

#pragma unroll
    for (int g = 0; g < 2; ++g) {
      v8f c[4];
#pragma unroll
      for (int jj = 0; jj < 4; ++jj) {
        const float bj = bj2[g * 4 + jj];
#pragma unroll
        for (int r = 0; r < 8; ++r) c[jj][r] = bj;
      }

#pragma unroll
      for (int s = 0; s < 4; ++s) {
#pragma unroll
        for (int jj = 0; jj < 4; ++jj) {
          const int colf = (g * 4 + jj) * 16 + n;
          const uint4* pb = reinterpret_cast<const uint4*>(w2b_l + colf * HDIM + half * 8);
          v16bf b = make_frag(pb[s * 4], pb[s * 4 + 2]);
          c[jj] = __builtin_amdgcn_wmma_f32_16x16x32_bf16(false, a2[s], false, b,
                                                          (short)0, c[jj], false, false);
        }
      }
#pragma unroll
      for (int jj = 0; jj < 4; ++jj) {
        const float wj = w3v[g * 4 + jj];
#pragma unroll
        for (int r = 0; r < 8; ++r) p[r] = __builtin_fmaf(silu(c[jj][r]), wj, p[r]);
      }
    }

    // Reduce over the 16 N-lanes of each half: every lane gets phi[r + 8*half]
#pragma unroll
    for (int mask = 1; mask <= 8; mask <<= 1) {
#pragma unroll
      for (int r = 0; r < 8; ++r) p[r] += __shfl_xor(p[r], mask, 32);
    }

    // --- scatter: trans = coord_diff * phi * edge_mask / 100 into out[row] ---
    if (n < 3) {
#pragma unroll
      for (int r = 0; r < 8; ++r) {
        const long long m = e0 + 8 * half + r;
        if (m < E) {
          const int dst  = ei[m];                       // row index
          const float em = edge_mask[m];
          const float tr = coord_diff[m * 3 + n] * p[r] * em * INV_NORM;
          atomicAdd(&out[(long long)dst * 3 + n], tr);
        }
      }
    }
  }
}

// out *= node_mask
__global__ void finalize_kernel(float* __restrict__ out, const float* __restrict__ node_mask,
                                int N3) {
  int t = blockIdx.x * blockDim.x + threadIdx.x;
  if (t < N3) out[t] *= node_mask[t / 3];
}

// ---------------------------------------------------------------------------
extern "C" void kernel_launch(void* const* d_in, const int* in_sizes, int n_in,
                              void* d_out, int out_size, void* d_ws, size_t ws_size,
                              hipStream_t stream) {
  const float* h          = (const float*)d_in[0];
  const float* coord      = (const float*)d_in[1];
  const int*   ei         = (const int*)d_in[2];
  const float* coord_diff = (const float*)d_in[3];
  const float* edge_attr  = (const float*)d_in[4];
  const float* node_mask  = (const float*)d_in[5];
  const float* edge_mask  = (const float*)d_in[6];
  const float* W1         = (const float*)d_in[7];
  const float* b1         = (const float*)d_in[8];
  const float* W2         = (const float*)d_in[9];
  const float* b2         = (const float*)d_in[10];
  const float* W3         = (const float*)d_in[11];
  float* out = (float*)d_out;

  const int N = in_sizes[1] / 3;      // coord is [N,3]
  const int E = in_sizes[4];          // edge_attr is [E,1]

  char* ws = (char*)d_ws;
  unsigned short* w1a = (unsigned short*)(ws);                 // 32 KB
  unsigned short* w1b = (unsigned short*)(ws + 32768);         // 32 KB
  unsigned short* w2b = (unsigned short*)(ws + 65536);         // 32 KB
  unsigned short* h16 = (unsigned short*)(ws + 98304);         // N*128*2 B

  prep_weights<<<(HDIM * HDIM + 255) / 256, 256, 0, stream>>>(W1, W2, w1a, w1b, w2b);

  const int nh = N * HDIM;
  prep_nodes<<<(nh + 255) / 256, 256, 0, stream>>>(h, coord, h16, out, N);

  const long long tiles  = ((long long)E + 15) / 16;
  const long long perblk = 8LL * TILES_PER_WAVE;               // tiles per block
  const int blocks = (int)((tiles + perblk - 1) / perblk);
  edge_kernel<<<blocks, 256, 0, stream>>>(h16, (const uint4*)ws, ei,
                                          coord_diff, edge_attr, edge_mask,
                                          W1, b1, b2, W3, out, E);

  finalize_kernel<<<(N * 3 + 255) / 256, 256, 0, stream>>>(out, node_mask, N * 3);
}
